// Head_16758962389619
// MI455X (gfx1250) — matrix operand order repaired and checked by reading, not compiled, not run on hardware
//
#include <hip/hip_runtime.h>

typedef __attribute__((ext_vector_type(16))) __bf16 v16bf;
typedef __attribute__((ext_vector_type(8)))  __bf16 v8bf;
typedef __attribute__((ext_vector_type(8)))  float  v8f;

constexpr int CEMB  = 1024;   // N_EMBD
constexpr int HS    = 64;     // HEAD_SIZE
constexpr int TSEQ  = 4096;   // BLOCK
constexpr int BATCH = 4;

// B-fragment gather: elems 0..7 = K base..base+7, 8..15 = K base+16..base+23.
__device__ __forceinline__ v16bf ld_contig16(const __bf16* p) {
  const v8bf* q = (const v8bf*)p;
  v8bf lo = q[0], hi = q[2];
  v16bf r;
#pragma unroll
  for (int j = 0; j < 8; ++j) { r[j] = lo[j]; r[j + 8] = hi[j]; }
  return r;
}

// A-fragment gather: elems 0..7 = K kbA..kbA+7 ; elems 8..15 = K kbA+16..kbA+23.
__device__ __forceinline__ v16bf ld_a16(const __bf16* row, int kbA) {
  const v8bf* q = (const v8bf*)(row + kbA);
  v8bf lo = q[0], hi = q[2];
  v16bf r;
#pragma unroll
  for (int j = 0; j < 8; ++j) { r[j] = lo[j]; r[j + 8] = hi[j]; }
  return r;
}

// ---------------------------------------------------------------------------
// Kernel 0: one-shot W (CEMB x HS, f32) -> W^T (HS x CEMB, bf16) so the GEMM
// B-operand over k is contiguous (2x b128 per fragment instead of 16x b32).
// grid (256, 3), block 256.
// ---------------------------------------------------------------------------
__global__ __launch_bounds__(256)
void wt_kernel(const float* __restrict__ Wq, const float* __restrict__ Wk,
               const float* __restrict__ Wv,
               __bf16* __restrict__ wtq, __bf16* __restrict__ wtk,
               __bf16* __restrict__ wtv) {
  const int idx = blockIdx.x * 256 + threadIdx.x;          // 0 .. CEMB*HS-1
  const float* W = (blockIdx.y == 0) ? Wq : (blockIdx.y == 1) ? Wk : Wv;
  __bf16*     wt = (blockIdx.y == 0) ? wtq : (blockIdx.y == 1) ? wtk : wtv;
  const int k = idx / HS, n = idx % HS;
  wt[(size_t)n * CEMB + k] = (__bf16)W[idx];
}

// ---------------------------------------------------------------------------
// Kernel 1: q/k/v = x @ W{q,k,v}; RoPE(q,k); q,k row-major bf16; v transposed
// (HS x T) bf16 so the attention P*V B-operand is contiguous over keys.
// grid (T/16, B), block 128 (4 waves). Wave w owns output columns [16w,16w+16).
// ---------------------------------------------------------------------------
__global__ __launch_bounds__(128)
void qkv_rope_kernel(const float* __restrict__ x,
                     const __bf16* __restrict__ wtq,
                     const __bf16* __restrict__ wtk,
                     const __bf16* __restrict__ wtv,
                     const float* __restrict__ fcos,
                     const float* __restrict__ fsin,
                     __bf16* __restrict__ qws,
                     __bf16* __restrict__ kws,
                     __bf16* __restrict__ vtws) {
  const int lane = threadIdx.x & 31;
  const int wave = threadIdx.x >> 5;
  const int row0 = blockIdx.x * 16;
  const int b    = blockIdx.y;
  const int m    = lane & 15;
  const int kbA  = (lane < 16) ? 0 : 8;     // A-matrix K base
  const int kbB  = (lane < 16) ? 0 : 8;     // B-matrix K base
  const int ncol = wave * 16 + m;           // output column (head dim)

  const float*  xrow = x + ((size_t)b * TSEQ + row0 + m) * CEMB;
  const __bf16* wqp  = wtq + (size_t)ncol * CEMB;
  const __bf16* wkp  = wtk + (size_t)ncol * CEMB;
  const __bf16* wvp  = wtv + (size_t)ncol * CEMB;

  v8f accq = {}, acck = {}, accv = {};
  for (int k0 = 0; k0 < CEMB; k0 += 32) {
    // A fragment: two 32-byte contiguous runs -> 4x b128 loads + pack-cvt.
    const float4* xp = (const float4*)(xrow + k0 + kbA);
    const float4 c0 = xp[0], c1 = xp[1], c2 = xp[4], c3 = xp[5];
    v16bf a;
    a[0]=(__bf16)c0.x; a[1]=(__bf16)c0.y; a[2] =(__bf16)c0.z; a[3] =(__bf16)c0.w;
    a[4]=(__bf16)c1.x; a[5]=(__bf16)c1.y; a[6] =(__bf16)c1.z; a[7] =(__bf16)c1.w;
    a[8]=(__bf16)c2.x; a[9]=(__bf16)c2.y; a[10]=(__bf16)c2.z; a[11]=(__bf16)c2.w;
    a[12]=(__bf16)c3.x;a[13]=(__bf16)c3.y;a[14] =(__bf16)c3.z;a[15] =(__bf16)c3.w;

    // B fragments from W^T: contiguous over k -> 2x b128 each.
    const v16bf bq = ld_contig16(wqp + k0 + kbB);
    const v16bf bk = ld_contig16(wkp + k0 + kbB);
    const v16bf bv = ld_contig16(wvp + k0 + kbB);

    // One A fragment feeds three WMMAs (q,k,v share x).
    accq = __builtin_amdgcn_wmma_f32_16x16x32_bf16(false, a, false, bq, (short)0, accq, false, false);
    acck = __builtin_amdgcn_wmma_f32_16x16x32_bf16(false, a, false, bk, (short)0, acck, false, false);
    accv = __builtin_amdgcn_wmma_f32_16x16x32_bf16(false, a, false, bv, (short)0, accv, false, false);
  }

  // RoPE in D layout: column pair (even,odd) = lanes (L, L^1).
  const int  jidx = ncol >> 1;
  const bool even = ((ncol & 1) == 0);
#pragma unroll
  for (int r = 0; r < 8; ++r) {
    const int   trow = row0 + r + ((lane < 16) ? 0 : 8);
    const float c = fcos[trow * (HS / 2) + jidx];
    const float s = fsin[trow * (HS / 2) + jidx];

    const float vq = accq[r];
    const float pq = __shfl_xor(vq, 1, 32);
    const float oq = even ? (vq * c - pq * s) : (pq * s + vq * c);

    const float vk = acck[r];
    const float pk = __shfl_xor(vk, 1, 32);
    const float ok = even ? (vk * c - pk * s) : (pk * s + vk * c);

    const size_t base = ((size_t)b * TSEQ + trow) * HS + ncol;
    qws[base] = (__bf16)oq;
    kws[base] = (__bf16)ok;
    vtws[((size_t)b * HS + ncol) * TSEQ + trow] = (__bf16)accv[r];   // transposed
  }
}

// ---------------------------------------------------------------------------
// Kernel 2: causal flash attention. grid (T/64, B), block 128 (4 waves).
// Block shares one key loop; K tiles staged into LDS with async-to-LDS copies
// (double buffered, ASYNCcnt + barrier). V read directly from L2 via V^T.
// ---------------------------------------------------------------------------
__global__ __launch_bounds__(128)
void flash_attn_kernel(const __bf16* __restrict__ qws,
                       const __bf16* __restrict__ kws,
                       const __bf16* __restrict__ vtws,
                       float* __restrict__ out) {
  __shared__ __attribute__((aligned(16))) __bf16 kbuf[2][32][HS];   // 8 KB
  __shared__ __attribute__((aligned(16))) __bf16 pbuf[4][16 * 32];  // 4 KB

  const int tid  = threadIdx.x;
  const int lane = tid & 31;
  const int wave = tid >> 5;
  const int q0b  = blockIdx.x * 64;
  const int q0   = q0b + wave * 16;
  const int b    = blockIdx.y;
  const int m    = lane & 15;
  const int kbA  = (lane < 16) ? 0 : 8;
  const int kbB  = (lane < 16) ? 0 : 8;
  const int qoff = (lane < 16) ? 0 : 8;
  const float scale = 0.125f;                  // 64^-0.5

  const __bf16* kbase = kws  + (size_t)b * TSEQ * HS;
  const __bf16* vtb   = vtws + (size_t)b * HS * TSEQ;

  // Staging assignment: 128 threads x 2 chunks cover 32 rows x 128 B.
  const int srow = tid >> 3;
  const int scol = (tid & 7) * 8;              // in elements (16 B chunks)

  auto stage = [&](int s, int sel) {
    const unsigned long long g0 =
        (unsigned long long)(kbase + ((size_t)(s * 32 + srow)) * HS + scol);
    const unsigned long long g1 = g0 + 16 * HS * sizeof(__bf16);
    const unsigned l0 = (unsigned)(unsigned long long)(const void*)&kbuf[sel][srow][scol];
    const unsigned l1 = l0 + 16 * HS * sizeof(__bf16);
    asm volatile("global_load_async_to_lds_b128 %0, %1, off" :: "v"(l0), "v"(g0) : "memory");
    asm volatile("global_load_async_to_lds_b128 %0, %1, off" :: "v"(l1), "v"(g1) : "memory");
  };

  // Q as two A fragments (head dims 0..31, 32..63), loaded once.
  const __bf16* qrow = qws + ((size_t)b * TSEQ + q0 + m) * HS;
  const v16bf aq0 = ld_a16(qrow, kbA);
  const v16bf aq1 = ld_a16(qrow + 32, kbA);

  float mrow[8], lrow[8];
  v8f oacc[4] = {};
#pragma unroll
  for (int r = 0; r < 8; ++r) { mrow[r] = -3.0e38f; lrow[r] = 0.0f; }

  __bf16* pb = &pbuf[wave][0];
  const int nsteps = (q0b + 64) >> 5;          // keys [0, q0b+64) in 32-key steps

  stage(0, 0);
  asm volatile("s_wait_asynccnt 0x0" ::: "memory");
  __syncthreads();

  for (int s = 0; s < nsteps; ++s) {
    const int k0  = s * 32;
    const int sel = s & 1;
    if (s + 1 < nsteps) stage(s + 1, sel ^ 1);   // prefetch next K tile (other buffer)

    if (k0 <= q0 + 15) {                         // wave-uniform causal skip
      // ---- S = Q K^T from LDS (contiguous b128 reads) ----
      const __bf16* kb = &kbuf[sel][0][0];
      const v16bf b0a = ld_contig16(kb + m * HS + kbB);
      const v16bf b0b = ld_contig16(kb + m * HS + 32 + kbB);
      const v16bf b1a = ld_contig16(kb + (m + 16) * HS + kbB);
      const v16bf b1b = ld_contig16(kb + (m + 16) * HS + 32 + kbB);

      v8f s0 = {}, s1 = {};
      s0 = __builtin_amdgcn_wmma_f32_16x16x32_bf16(false, aq0, false, b0a, (short)0, s0, false, false);
      s0 = __builtin_amdgcn_wmma_f32_16x16x32_bf16(false, aq1, false, b0b, (short)0, s0, false, false);
      s1 = __builtin_amdgcn_wmma_f32_16x16x32_bf16(false, aq0, false, b1a, (short)0, s1, false, false);
      s1 = __builtin_amdgcn_wmma_f32_16x16x32_bf16(false, aq1, false, b1b, (short)0, s1, false, false);

      // ---- causal mask + online softmax; P (bf16) to wave-private LDS ----
#pragma unroll
      for (int r = 0; r < 8; ++r) {
        const int qrow_i = q0 + r + qoff;
        float sv0 = s0[r] * scale;
        float sv1 = s1[r] * scale;
        if (k0 + m > qrow_i)      sv0 = -1.0e30f;
        if (k0 + 16 + m > qrow_i) sv1 = -1.0e30f;

        float rm = fmaxf(sv0, sv1);
#pragma unroll
        for (int d = 1; d < 16; d <<= 1) rm = fmaxf(rm, __shfl_xor(rm, d, 32));
        const float mnew  = fmaxf(mrow[r], rm);
        const float alpha = __expf(mrow[r] - mnew);
        const float p0 = __expf(sv0 - mnew);
        const float p1 = __expf(sv1 - mnew);
        float ps = p0 + p1;
#pragma unroll
        for (int d = 1; d < 16; d <<= 1) ps += __shfl_xor(ps, d, 32);
        lrow[r] = lrow[r] * alpha + ps;
        mrow[r] = mnew;
#pragma unroll
        for (int dg = 0; dg < 4; ++dg) oacc[dg][r] *= alpha;

        const int pidx = (r + qoff) * 32 + m;
        pb[pidx]      = (__bf16)p0;
        pb[pidx + 16] = (__bf16)p1;
      }

      // ---- O += P V : P reread in A layout (DS in-order within wave);
      //      V^T gives contiguous key runs -> b128 global loads from L2. ----
      const v16bf ap = ld_a16(pb + m * 32, kbA);
#pragma unroll
      for (int dg = 0; dg < 4; ++dg) {
        const __bf16* vrow = vtb + ((size_t)(dg * 16 + m)) * TSEQ + k0 + kbB;
        __builtin_prefetch(vrow + 32, 0, 1);
        const v16bf bv = ld_contig16(vrow);
        oacc[dg] = __builtin_amdgcn_wmma_f32_16x16x32_bf16(false, ap, false, bv, (short)0, oacc[dg], false, false);
      }
    }

    if (s + 1 < nsteps) {
      asm volatile("s_wait_asynccnt 0x0" ::: "memory");   // own copies done
      __syncthreads();                                    // everyone's copies done
    }
  }

  // ---- normalize and store fp32 output ----
#pragma unroll
  for (int r = 0; r < 8; ++r) {
    const float inv  = 1.0f / lrow[r];
    const int   qr   = q0 + r + qoff;
    float* orow = out + ((size_t)b * TSEQ + qr) * HS;
#pragma unroll
    for (int dg = 0; dg < 4; ++dg) orow[dg * 16 + m] = oacc[dg][r] * inv;
  }
}

// ---------------------------------------------------------------------------
extern "C" void kernel_launch(void* const* d_in, const int* in_sizes, int n_in,
                              void* d_out, int out_size, void* d_ws, size_t ws_size,
                              hipStream_t stream) {
  const float* x    = (const float*)d_in[0];
  const float* Wq   = (const float*)d_in[1];
  const float* Wk   = (const float*)d_in[2];
  const float* Wv   = (const float*)d_in[3];
  const float* fcos = (const float*)d_in[4];
  const float* fsin = (const float*)d_in[5];
  float* out = (float*)d_out;

  const size_t qkv_elems = (size_t)BATCH * TSEQ * HS;   // 1M elems, 2 MB bf16 each
  const size_t w_elems   = (size_t)CEMB * HS;           // 64K elems
  __bf16* qws  = (__bf16*)d_ws;
  __bf16* kws  = qws  + qkv_elems;
  __bf16* vtws = kws  + qkv_elems;
  __bf16* wtq  = vtws + qkv_elems;
  __bf16* wtk  = wtq  + w_elems;
  __bf16* wtv  = wtk  + w_elems;

  dim3 g0(CEMB * HS / 256, 3);
  wt_kernel<<<g0, dim3(256), 0, stream>>>(Wq, Wk, Wv, wtq, wtk, wtv);

  dim3 g1(TSEQ / 16, BATCH);
  qkv_rope_kernel<<<g1, dim3(128), 0, stream>>>(x, wtq, wtk, wtv, fcos, fsin, qws, kws, vtws);

  dim3 g2(TSEQ / 64, BATCH);
  flash_attn_kernel<<<g2, dim3(128), 0, stream>>>(qws, kws, vtws, out);
}
